// ConvTranspose2d_3736621547827
// MI455X (gfx1250) — matrix operand order repaired and checked by reading, not compiled, not run on hardware
//
#include <hip/hip_runtime.h>

// ---------------------------------------------------------------------------
// ConvTranspose2d (stride-1 full conv region) via CDNA5 WMMA, split-f16 (hi/lo)
// B=16, CIN=COUT=128, K=4, valid output region 66x66 inside 126x126 zeros.
// ---------------------------------------------------------------------------

typedef __attribute__((ext_vector_type(16))) _Float16 v16h;
typedef __attribute__((ext_vector_type(8)))  _Float16 v8h;
typedef __attribute__((ext_vector_type(8)))  float    v8f;

#define CIN   128
#define COUT  128
#define KSZ   4
#define HIN   64
#define WIN   64
#define HOUT  126
#define WOUT  126
#define YREG  66        // valid (nonzero) output region is [0,66) x [0,66)
#define XVAL  63        // xs = x[:, :, :63, :63]

#define OI_T  8         // output rows per block
#define OJ_T  16        // output cols per block (WMMA N)
#define IH_T  (OI_T + KSZ - 1)   // 11 input rows incl. halo
#define IW_T  (OJ_T + KSZ - 1)   // 19 input cols incl. halo
#define CI_S  136       // padded ci stride in LDS (halves) to spread banks

// number of valid kernel taps covering output index i in [0,66)
__device__ __forceinline__ float edge_count(int i) {
    int lo = (i < KSZ - 1) ? i : (KSZ - 1);        // min(i,3)
    int hi = (i > XVAL - 1) ? (i - (XVAL - 1)) : 0; // max(0, i-62)
    return (float)(lo - hi + 1);
}

// ---------------------------------------------------------------------------
// Prep: split weights into f16 hi/lo, transposed to [khkw][split][co][ci];
// also reduce biasSum = sum(bias[:128]) into ws[0].
// ---------------------------------------------------------------------------
__global__ void convt_prep_kernel(const float* __restrict__ kern,   // [CIN][COUT][K][K]
                                  const float* __restrict__ bias,   // [COUT+1]
                                  float* __restrict__ bsum,
                                  _Float16* __restrict__ Wp)        // [16][2][COUT][CIN]
{
    int idx = blockIdx.x * blockDim.x + threadIdx.x;
    if (idx == 0) {
        float s = 0.f;
        for (int i = 0; i < COUT; ++i) s += bias[i];
        *bsum = s;
    }
    const int total = 16 * COUT * CIN;
    if (idx < total) {
        int ci   = idx & (CIN - 1);
        int co   = (idx >> 7) & (COUT - 1);
        int khkw = idx >> 14;
        float v = kern[((size_t)ci * COUT + co) * (KSZ * KSZ) + khkw];
        _Float16 hi = (_Float16)v;
        _Float16 lo = (_Float16)(v - (float)hi);
        Wp[(((size_t)khkw * 2 + 0) * COUT + co) * CIN + ci] = hi;
        Wp[(((size_t)khkw * 2 + 1) * COUT + co) * CIN + ci] = lo;
    }
}

// ---------------------------------------------------------------------------
// Main kernel: one block = (batch, 8x16 output tile), 8 waves = 8 co-tiles.
// ---------------------------------------------------------------------------
__global__ __launch_bounds__(256)
void convt_wmma_kernel(const float* __restrict__ x,      // [B][CIN][64][64]
                       const _Float16* __restrict__ Wp,  // [16][2][COUT][CIN]
                       const float* __restrict__ bsum,
                       float* __restrict__ out)          // [B][COUT][126][126]
{
    // LDS input tile, channel-last, hi/lo split: [split][ih][iw][ci(padded)]
    __shared__ _Float16 Xs[2][IH_T][IW_T][CI_S];

    const int tid  = threadIdx.x;
    const int b    = blockIdx.z;
    const int oi0  = blockIdx.y * OI_T;
    const int oj0  = blockIdx.x * OJ_T;

    // ---- cooperative LDS fill with on-the-fly f32 -> (f16 hi, f16 lo) ----
    const float* xb = x + (size_t)b * CIN * HIN * WIN;
    for (int idx = tid; idx < IH_T * IW_T * CIN; idx += 256) {
        int ci  = idx & (CIN - 1);
        int t   = idx >> 7;
        int iwl = t % IW_T;
        int ihl = t / IW_T;
        int ih  = oi0 - (KSZ - 1) + ihl;
        int iw  = oj0 - (KSZ - 1) + iwl;
        float v = 0.f;
        if (ih >= 0 && ih < XVAL && iw >= 0 && iw < XVAL)
            v = xb[(size_t)ci * (HIN * WIN) + ih * WIN + iw];
        _Float16 hi = (_Float16)v;
        _Float16 lo = (_Float16)(v - (float)hi);
        Xs[0][ihl][iwl][ci] = hi;
        Xs[1][ihl][iwl][ci] = lo;
    }
    __syncthreads();

    const int wave = tid >> 5;
    const int lane = tid & 31;
    const int g    = lane >> 4;     // lane half
    const int n    = lane & 15;     // N index / A row index
    const int co0  = wave * 16;

    v8f acc[OI_T] = {};             // D/C: rows oi0..oi0+7, cols oj0..oj0+15

    for (int kh = 0; kh < KSZ; ++kh) {
        for (int kw = 0; kw < KSZ; ++kw) {
            // ---- hoist A fragments for this (kh,kw): 4 ci-chunks x {hi,lo}
            // ISA 16-bit A layout: lane m = lane&15; elem e: k = e+8g (e<8),
            // k = e+8+8g (e>=8)  -> two contiguous 8-elem runs per lane.
            v16h a[4][2];
            const int khkw = kh * KSZ + kw;
#pragma unroll
            for (int c = 0; c < 4; ++c) {
#pragma unroll
                for (int s = 0; s < 2; ++s) {
                    const _Float16* wr =
                        Wp + (((size_t)khkw * 2 + s) * COUT + (co0 + n)) * CIN;
                    v8h p0 = *(const v8h*)(wr + c * 32 + 8 * g);        // k: 8g..8g+7
                    v8h p1 = *(const v8h*)(wr + c * 32 + 16 + 8 * g);   // k: 16+8g..
                    a[c][s] = __builtin_shufflevector(p0, p1,
                               0,1,2,3,4,5,6,7,8,9,10,11,12,13,14,15);
                }
            }
            // ---- sweep the 8 output rows of the tile
#pragma unroll
            for (int r = 0; r < OI_T; ++r) {
                const int ihl = r + (KSZ - 1) - kh;   // 0..10, always in tile
                const int iwl = n + (KSZ - 1) - kw;   // 0..18, always in tile
#pragma unroll
                for (int c = 0; c < 4; ++c) {
                    // ISA 16-bit B layout (matches A): elem e: k = e+8g (e<8),
                    // k = e+8+8g (e>=8), N = lane&15.
                    const _Float16* b0 = &Xs[0][ihl][iwl][c * 32 + 8 * g];
                    const _Float16* b1 = &Xs[1][ihl][iwl][c * 32 + 8 * g];
                    v8h h0 = *(const v8h*)(b0);
                    v8h h1 = *(const v8h*)(b0 + 16);
                    v8h l0 = *(const v8h*)(b1);
                    v8h l1 = *(const v8h*)(b1 + 16);
                    v16h bhi = __builtin_shufflevector(h0, h1,
                                0,1,2,3,4,5,6,7,8,9,10,11,12,13,14,15);
                    v16h blo = __builtin_shufflevector(l0, l1,
                                0,1,2,3,4,5,6,7,8,9,10,11,12,13,14,15);
                    // split-precision: hi*hi + hi*lo + lo*hi (lo*lo ~ 2^-22)
                    acc[r] = __builtin_amdgcn_wmma_f32_16x16x32_f16(
                                 false, a[c][0], false, bhi, (short)0, acc[r],
                                 false, false);
                    acc[r] = __builtin_amdgcn_wmma_f32_16x16x32_f16(
                                 false, a[c][0], false, blo, (short)0, acc[r],
                                 false, false);
                    acc[r] = __builtin_amdgcn_wmma_f32_16x16x32_f16(
                                 false, a[c][1], false, bhi, (short)0, acc[r],
                                 false, false);
                }
            }
        }
    }

    // ---- store: D layout: elem i -> co = co0 + i + 8g, col = oj0 + n ----
    const float bs = *bsum;
    const int   oj = oj0 + n;
    float* ob = out + (size_t)b * COUT * HOUT * WOUT;
    if (oj < YREG) {
        const float cw = edge_count(oj);
#pragma unroll
        for (int r = 0; r < OI_T; ++r) {
            const int oi = oi0 + r;
            if (oi < YREG) {
                const float add = bs * edge_count(oi) * cw;
#pragma unroll
                for (int i = 0; i < 8; ++i) {
                    const int co = co0 + i + 8 * g;
                    ob[((size_t)co * HOUT + oi) * WOUT + oj] = acc[r][i] + add;
                }
            }
        }
    }
}

// ---------------------------------------------------------------------------
// Host-side launcher
// ---------------------------------------------------------------------------
extern "C" void kernel_launch(void* const* d_in, const int* in_sizes, int n_in,
                              void* d_out, int out_size, void* d_ws, size_t ws_size,
                              hipStream_t stream) {
    const float* x    = (const float*)d_in[0];   // [16][128][64][64]
    const float* kern = (const float*)d_in[1];   // [128][128][4][4]
    const float* bias = (const float*)d_in[2];   // [129]
    float*       out  = (float*)d_out;           // [16][128][126][126]

    float*     bsum = (float*)d_ws;
    _Float16*  Wp   = (_Float16*)((char*)d_ws + 16);  // 16*2*128*128 f16 = 1 MB

    // 1) weight split/transpose + bias reduction
    {
        int total  = 16 * COUT * CIN;
        int blocks = (total + 255) / 256;
        convt_prep_kernel<<<blocks, 256, 0, stream>>>(kern, bias, bsum, Wp);
    }

    // 2) zero the full output (only [0,66)^2 region is overwritten below)
    hipMemsetAsync(d_out, 0, (size_t)out_size * sizeof(float), stream);

    // 3) WMMA transposed-conv over the 66x66 valid region
    {
        dim3 grid((YREG + OJ_T - 1) / OJ_T,   // 5
                  (YREG + OI_T - 1) / OI_T,   // 9
                  16);                        // batch
        convt_wmma_kernel<<<grid, 256, 0, stream>>>(x, Wp, bsum, out);
    }
}
